// PhaseSyncAttentionWSA_36661840839318
// MI455X (gfx1250) — matrix operand order repaired and checked by reading, not compiled, not run on hardware
//
#include <hip/hip_runtime.h>
#include <hip/hip_bf16.h>

typedef __attribute__((ext_vector_type(16))) _Float16 v16h;
typedef __attribute__((ext_vector_type(8)))  float    v8f;

#define B_   4
#define L_   2048
#define D_   1024
#define H_   16
#define NB_  8
#define HD_  64
#define BL_  (B_*L_)      /* 8192  */
#define BHL_ (B_*H_*L_)   /* 131072 */

// ---------------------------------------------------------------------------
// CDNA5 async global->LDS copy (ASYNCcnt-tracked), 16 bytes per lane.
// VDST = VGPR with LDS byte address, VADDR = 64-bit global address, saddr=off.
// ---------------------------------------------------------------------------
__device__ __forceinline__ void async_copy_b128(void* lds_dst, const void* gsrc) {
    unsigned int laddr = (unsigned int)(unsigned long long)lds_dst; // low 32b = LDS offset
    asm volatile("global_load_async_to_lds_b128 %0, %1, off"
                 :: "v"(laddr), "v"(gsrc)
                 : "memory");
}
__device__ __forceinline__ void wait_async_all() {
    asm volatile("s_wait_asynccnt 0" ::: "memory");
}

// ---------------------------------------------------------------------------
// lane reductions within 16-lane halves (wave32) via ds_swizzle (xor pattern)
// imm = xor<<10 | or<<5 | and ; and=0x1f, or=0
// ---------------------------------------------------------------------------
__device__ __forceinline__ float red_max16(float v) {
    v = fmaxf(v, __int_as_float(__builtin_amdgcn_ds_swizzle(__float_as_int(v), 0x041F)));
    v = fmaxf(v, __int_as_float(__builtin_amdgcn_ds_swizzle(__float_as_int(v), 0x081F)));
    v = fmaxf(v, __int_as_float(__builtin_amdgcn_ds_swizzle(__float_as_int(v), 0x101F)));
    v = fmaxf(v, __int_as_float(__builtin_amdgcn_ds_swizzle(__float_as_int(v), 0x201F)));
    return v;
}
__device__ __forceinline__ float red_add16(float v) {
    v += __int_as_float(__builtin_amdgcn_ds_swizzle(__float_as_int(v), 0x041F));
    v += __int_as_float(__builtin_amdgcn_ds_swizzle(__float_as_int(v), 0x081F));
    v += __int_as_float(__builtin_amdgcn_ds_swizzle(__float_as_int(v), 0x101F));
    v += __int_as_float(__builtin_amdgcn_ds_swizzle(__float_as_int(v), 0x201F));
    return v;
}

// ---------------------------------------------------------------------------
// f32 -> f16 elementwise convert (4 elems / thread)
// ---------------------------------------------------------------------------
__global__ void psa_cvt_f16(const float* __restrict__ src, _Float16* __restrict__ dst, int n) {
    int base = (blockIdx.x * blockDim.x + threadIdx.x) * 4;
    if (base + 3 < n) {
        float4 v = *(const float4*)(src + base);
        dst[base + 0] = (_Float16)v.x;
        dst[base + 1] = (_Float16)v.y;
        dst[base + 2] = (_Float16)v.z;
        dst[base + 3] = (_Float16)v.w;
    }
}

// ---------------------------------------------------------------------------
// weight convert + transpose: W[k][n] f32 (1024x1024) -> Wt[n][k] f16
// (one-time 4MB cost so GEMM B-tiles become pure async copies)
// ---------------------------------------------------------------------------
__global__ void psa_cvt_tr_f16(const float* __restrict__ W, _Float16* __restrict__ Wt) {
    int t = blockIdx.x * blockDim.x + threadIdx.x;      // 0 .. 1048576/4-1
    int k  = t >> 8;                                    // 1024/4 = 256 chunks per row
    int nc = (t & 255) * 4;
    float4 v = *(const float4*)(W + (size_t)k * D_ + nc);
    Wt[(size_t)(nc + 0) * D_ + k] = (_Float16)v.x;
    Wt[(size_t)(nc + 1) * D_ + k] = (_Float16)v.y;
    Wt[(size_t)(nc + 2) * D_ + k] = (_Float16)v.z;
    Wt[(size_t)(nc + 3) * D_ + k] = (_Float16)v.w;
}

// ---------------------------------------------------------------------------
// phase projections: [B,NB,L] x [NB,H] + bp -> per (b,h,l) scalars
// ---------------------------------------------------------------------------
__global__ void psa_phase(const float* __restrict__ cosp, const float* __restrict__ sinp,
                          const float* __restrict__ mag,  const float* __restrict__ Wp,
                          const float* __restrict__ bp,
                          float* __restrict__ pc, float* __restrict__ ps,
                          float* __restrict__ pmc, float* __restrict__ pms) {
    int t = blockIdx.x * blockDim.x + threadIdx.x;
    if (t >= BHL_) return;
    int l = t % L_;
    int h = (t / L_) % H_;
    int b = t / (H_ * L_);
    float sc = 0.f, ss = 0.f, sm = 0.f;
#pragma unroll
    for (int n = 0; n < NB_; ++n) {
        float w = Wp[n * H_ + h];
        int a = (b * NB_ + n) * L_ + l;
        sc += cosp[a] * w;
        ss += sinp[a] * w;
        sm += mag[a]  * w;
    }
    float bb = bp[h];
    sc += bb; ss += bb; sm += bb;
    pc[t]  = sc;
    ps[t]  = ss;
    pmc[t] = sm * sc;
    pms[t] = sm * ss;
}

// ---------------------------------------------------------------------------
// WMMA GEMM: C[8192,1024] = A[8192,1024] @ W, W given TRANSPOSED (Wt[n][k]).
//   mode 0: Q  -> (acc+bq)*0.125 f16, [B,H,L,HD]
//   mode 1: K  -> (acc+bk)       f16, [B,H,L,HD]
//   mode 2: V  -> V_real/V_imag  f16 (fold pmc/pms), HEAD-TRANSPOSED [B,H,HD,L]
//   mode 3: O  -> acc+bo+residual f32, [B,L,D]
// 256 thr = 8 waves (4Mx2N), block tile 128x128, wave tile 32x64,
// K-step 32, async LDS staging; 8 WMMA per wave per k-step.
// ---------------------------------------------------------------------------
__launch_bounds__(256)
__global__ void psa_gemm(const _Float16* __restrict__ A,
                         const _Float16* __restrict__ Wqt, const _Float16* __restrict__ Wkt,
                         const _Float16* __restrict__ Wvt, const _Float16* __restrict__ Wot,
                         const float* __restrict__ bq, const float* __restrict__ bk,
                         const float* __restrict__ bv, const float* __restrict__ bo,
                         const float* __restrict__ resid,
                         const float* __restrict__ pmc, const float* __restrict__ pms,
                         _Float16* __restrict__ Qh, _Float16* __restrict__ Kh,
                         _Float16* __restrict__ Vrh, _Float16* __restrict__ Vih,
                         float* __restrict__ ybuf, int mode_base) {
    __shared__ __align__(16) _Float16 As[128][40];    // A tile 128x32, pitch 40
    __shared__ __align__(16) _Float16 Bts[128][40];   // Wt tile [n][k] 128x32

    const int mode = mode_base + (int)blockIdx.z;
    const _Float16* Bw = (mode == 0) ? Wqt : (mode == 1) ? Wkt : (mode == 2) ? Wvt : Wot;

    const int tid  = threadIdx.x;
    const int lane = tid & 31;
    const int w    = tid >> 5;
    const int waveM = w >> 1;          // 0..3 -> 32 rows each
    const int waveN = w & 1;           // 0..1 -> 64 cols each
    const int m0 = blockIdx.y * 128;
    const int n0 = blockIdx.x * 128;

    // per-thread staging coordinates: 512 chunks (16B) per 128x32 tile
    const int sr0 = tid >> 2,           sc0 = (tid & 3);
    const int sr1 = (tid + 256) >> 2,   sc1 = ((tid + 256) & 3);

    v8f acc[2][4];
    const v8f vzero = (v8f)0.0f;
#pragma unroll
    for (int i = 0; i < 2; ++i)
#pragma unroll
        for (int j = 0; j < 4; ++j) acc[i][j] = vzero;

    for (int k0 = 0; k0 < D_; k0 += 32) {
        // async stage: A tile + B tile, pure 16B copies (2 + 2 per thread)
        async_copy_b128(&As[sr0][sc0 * 8],  A  + (size_t)(m0 + sr0) * D_ + k0 + sc0 * 8);
        async_copy_b128(&As[sr1][sc1 * 8],  A  + (size_t)(m0 + sr1) * D_ + k0 + sc1 * 8);
        async_copy_b128(&Bts[sr0][sc0 * 8], Bw + (size_t)(n0 + sr0) * D_ + k0 + sc0 * 8);
        async_copy_b128(&Bts[sr1][sc1 * 8], Bw + (size_t)(n0 + sr1) * D_ + k0 + sc1 * 8);
        if (k0 + 32 < D_) {
            __builtin_prefetch(A  + (size_t)(m0 + sr0) * D_ + k0 + 32, 0, 3);
            __builtin_prefetch(Bw + (size_t)(n0 + sr0) * D_ + k0 + 32, 0, 3);
        }
        wait_async_all();
        __syncthreads();

        const int lm = lane & 15;
        const int ko = (lane < 16) ? 0 : 8;    // unused; A frags also use kb
        const int kb = (lane < 16) ? 0 : 16;   // B frag K split

        v16h af[2];
#pragma unroll
        for (int mi = 0; mi < 2; ++mi) {
            const _Float16* p = &As[waveM * 32 + mi * 16 + lm][0];
            union { v16h v; uint4 q[2]; } u;
            u.q[0] = *(const uint4*)(p + kb);
            u.q[1] = *(const uint4*)(p + kb + 8);
            af[mi] = u.v;
        }
#pragma unroll
        for (int ni = 0; ni < 4; ++ni) {
            const _Float16* p = &Bts[waveN * 64 + ni * 16 + lm][0];
            union { v16h v; uint4 q[2]; } u;
            u.q[0] = *(const uint4*)(p + kb);
            u.q[1] = *(const uint4*)(p + kb + 8);
            v16h bf = u.v;
#pragma unroll
            for (int mi = 0; mi < 2; ++mi)
                acc[mi][ni] = __builtin_amdgcn_wmma_f32_16x16x32_f16(
                    false, af[mi], false, bf, (short)0, acc[mi][ni], false, false);
        }
        __syncthreads();
    }

    // epilogue: C layout: lane = N col, VGPR vr = M row (lanes 16-31 -> +8)
    const int lm   = lane & 15;
    const int half = lane >> 4;
#pragma unroll
    for (int mi = 0; mi < 2; ++mi) {
#pragma unroll
        for (int ni = 0; ni < 4; ++ni) {
            const int nn = n0 + waveN * 64 + ni * 16 + lm;
#pragma unroll
            for (int vr = 0; vr < 8; ++vr) {
                const int mm = m0 + waveM * 32 + mi * 16 + vr + half * 8;
                float v = acc[mi][ni][vr];
                const int b  = mm >> 11, l = mm & (L_ - 1);
                const int h  = nn >> 6,  hd = nn & (HD_ - 1);
                const size_t i3  = ((size_t)(b * H_ + h) * L_ + l) * HD_ + hd;   // [B,H,L,HD]
                const size_t i3v = ((size_t)(b * H_ + h) * HD_ + hd) * L_ + l;   // [B,H,HD,L]
                const size_t iph = (size_t)(b * H_ + h) * L_ + l;
                if (mode == 0) {
                    Qh[i3] = (_Float16)((v + bq[nn]) * 0.125f);   // fold 1/sqrt(HD)
                } else if (mode == 1) {
                    Kh[i3] = (_Float16)(v + bk[nn]);
                } else if (mode == 2) {
                    float vv = v + bv[nn];
                    Vrh[i3v] = (_Float16)(vv * pmc[iph]);
                    Vih[i3v] = (_Float16)(vv * pms[iph]);
                } else {
                    ybuf[(size_t)mm * D_ + nn] = v + bo[nn] + resid[(size_t)mm * D_ + nn];
                }
            }
        }
    }
}

// ---------------------------------------------------------------------------
// Flash attention with phase-split values.
// 256 thr = 8 waves; q-tile 128 (16 rows/wave), k-tile 64.
// K in [B,H,L,HD]; V_real/V_imag pre-transposed [B,H,HD,L] -> all tiles are
// pure copies staged with global_load_async_to_lds_b128.
// ---------------------------------------------------------------------------
__launch_bounds__(256)
__global__ void psa_attn(const _Float16* __restrict__ Qh, const _Float16* __restrict__ Kh,
                         const _Float16* __restrict__ Vrh, const _Float16* __restrict__ Vih,
                         const float* __restrict__ mask,
                         const float* __restrict__ pc, const float* __restrict__ ps,
                         _Float16* __restrict__ ctxh) {
    __shared__ __align__(16) _Float16 Ks[64][72];        // [key][hd]
    __shared__ __align__(16) _Float16 Vrt[64][72];       // [hd][key]
    __shared__ __align__(16) _Float16 Vit[64][72];       // [hd][key]
    __shared__ __align__(16) _Float16 Ps[8][16][72];     // per-wave P tile [qrow][key]

    const int tid  = threadIdx.x;
    const int lane = tid & 31;
    const int w    = tid >> 5;
    const int bh   = blockIdx.y;          // b*H + h
    const int b    = bh >> 4;
    const int h    = bh & 15;
    const int q0   = blockIdx.x * 128;

    const int lm   = lane & 15;
    const int half = lane >> 4;
    const int ko   = half ? 8 : 0;
    const int kb   = half ? 16 : 0;

    // Q fragments for this wave's 16 rows (registers)
    const size_t qbase = ((size_t)bh * L_ + q0 + w * 16 + lm) * HD_;
    v16h qf[2];
#pragma unroll
    for (int s = 0; s < 2; ++s) {
        union { v16h v; uint4 q[2]; } u;
        u.q[0] = *(const uint4*)(Qh + qbase + s * 32 + kb);
        u.q[1] = *(const uint4*)(Qh + qbase + s * 32 + kb + 8);
        qf[s] = u.v;
    }

    v8f cr[4], ci[4];
    const v8f vzero = (v8f)0.0f;
#pragma unroll
    for (int dt = 0; dt < 4; ++dt) { cr[dt] = vzero; ci[dt] = vzero; }
    float rmax[8], rsum[8];
#pragma unroll
    for (int vr = 0; vr < 8; ++vr) { rmax[vr] = -1e30f; rsum[vr] = 0.f; }

    // staging coordinates: 512 x 16B chunks per 64x64 tile -> 2 per thread
    const int r0 = (tid * 2 + 0) >> 3, c0 = (tid * 2 + 0) & 7;
    const int r1 = (tid * 2 + 1) >> 3, c1 = (tid * 2 + 1) & 7;

    for (int kb0 = 0; kb0 < L_; kb0 += 64) {
        // K tile: [key][hd], natural copy
        async_copy_b128(&Ks[r0][c0 * 8], Kh + ((size_t)bh * L_ + kb0 + r0) * HD_ + c0 * 8);
        async_copy_b128(&Ks[r1][c1 * 8], Kh + ((size_t)bh * L_ + kb0 + r1) * HD_ + c1 * 8);
        // V^T tiles: [hd][key], natural copy from [B,H,HD,L]
        async_copy_b128(&Vrt[r0][c0 * 8], Vrh + ((size_t)bh * HD_ + r0) * L_ + kb0 + c0 * 8);
        async_copy_b128(&Vrt[r1][c1 * 8], Vrh + ((size_t)bh * HD_ + r1) * L_ + kb0 + c1 * 8);
        async_copy_b128(&Vit[r0][c0 * 8], Vih + ((size_t)bh * HD_ + r0) * L_ + kb0 + c0 * 8);
        async_copy_b128(&Vit[r1][c1 * 8], Vih + ((size_t)bh * HD_ + r1) * L_ + kb0 + c1 * 8);
        if (kb0 + 64 < L_) {
            __builtin_prefetch(Kh  + ((size_t)bh * L_ + kb0 + 64 + r0) * HD_, 0, 3);
            __builtin_prefetch(Vrh + ((size_t)bh * HD_ + r0) * L_ + kb0 + 64, 0, 3);
            __builtin_prefetch(Vih + ((size_t)bh * HD_ + r0) * L_ + kb0 + 64, 0, 3);
        }
        wait_async_all();
        __syncthreads();

        // ---- S = Q * Kt  (4 key sub-tiles x 2 k-steps over HD) ----
        v8f sacc[4];
#pragma unroll
        for (int nt = 0; nt < 4; ++nt) {
            sacc[nt] = vzero;
#pragma unroll
            for (int s = 0; s < 2; ++s) {
                const _Float16* p = &Ks[nt * 16 + lm][s * 32 + kb];
                union { v16h v; uint4 q[2]; } u;
                u.q[0] = *(const uint4*)(p);
                u.q[1] = *(const uint4*)(p + 8);
                sacc[nt] = __builtin_amdgcn_wmma_f32_16x16x32_f16(
                    false, qf[s], false, u.v, (short)0, sacc[nt], false, false);
            }
        }
        // additive attention mask (per key column)
#pragma unroll
        for (int nt = 0; nt < 4; ++nt) {
            float mk = mask[(size_t)b * L_ + kb0 + nt * 16 + lm];
#pragma unroll
            for (int vr = 0; vr < 8; ++vr) sacc[nt][vr] += mk;
        }

        // ---- online softmax per owned row ----
#pragma unroll
        for (int vr = 0; vr < 8; ++vr) {
            float tmax = fmaxf(fmaxf(sacc[0][vr], sacc[1][vr]),
                               fmaxf(sacc[2][vr], sacc[3][vr]));
            tmax = red_max16(tmax);
            float nm    = fmaxf(rmax[vr], tmax);
            float scale = __expf(rmax[vr] - nm);
            rmax[vr]    = nm;
            float psum  = 0.f;
            const int prow = vr + half * 8;
#pragma unroll
            for (int nt = 0; nt < 4; ++nt) {
                float p = __expf(sacc[nt][vr] - nm);
                psum += p;
                Ps[w][prow][nt * 16 + lm] = (_Float16)p;
            }
            rsum[vr] = rsum[vr] * scale + psum;   // per-lane partial (4 cols)
#pragma unroll
            for (int dt = 0; dt < 4; ++dt) { cr[dt][vr] *= scale; ci[dt][vr] *= scale; }
        }

        // ---- ctx += P * V_real / V_imag ----
#pragma unroll
        for (int s = 0; s < 2; ++s) {
            const _Float16* pa = &Ps[w][lm][0];
            union { v16h v; uint4 q[2]; } ua;
            ua.q[0] = *(const uint4*)(pa + s * 32 + kb);
            ua.q[1] = *(const uint4*)(pa + s * 32 + kb + 8);
            v16h af = ua.v;
#pragma unroll
            for (int dt = 0; dt < 4; ++dt) {
                const _Float16* pr = &Vrt[dt * 16 + lm][s * 32 + kb];
                union { v16h v; uint4 q[2]; } ur;
                ur.q[0] = *(const uint4*)(pr);
                ur.q[1] = *(const uint4*)(pr + 8);
                cr[dt] = __builtin_amdgcn_wmma_f32_16x16x32_f16(
                    false, af, false, ur.v, (short)0, cr[dt], false, false);
                const _Float16* pi = &Vit[dt * 16 + lm][s * 32 + kb];
                union { v16h v; uint4 q[2]; } ui;
                ui.q[0] = *(const uint4*)(pi);
                ui.q[1] = *(const uint4*)(pi + 8);
                ci[dt] = __builtin_amdgcn_wmma_f32_16x16x32_f16(
                    false, af, false, ui.v, (short)0, ci[dt], false, false);
            }
        }
        __syncthreads();
    }

    // ---- finalize: normalize, apply query-side phases, store f16 ctx [B,L,H,HD] ----
    float inv[8], pcv[8], psv[8];
#pragma unroll
    for (int vr = 0; vr < 8; ++vr) {
        float s = red_add16(rsum[vr]);
        inv[vr] = 1.0f / s;
        const int row = q0 + w * 16 + vr + half * 8;
        const size_t ip = (size_t)bh * L_ + row;
        pcv[vr] = pc[ip];
        psv[vr] = ps[ip];
    }
#pragma unroll
    for (int dt = 0; dt < 4; ++dt) {
#pragma unroll
        for (int vr = 0; vr < 8; ++vr) {
            const int row = q0 + w * 16 + vr + half * 8;
            float o = (cr[dt][vr] * pcv[vr] + ci[dt][vr] * psv[vr]) * inv[vr];
            ctxh[(((size_t)b * L_ + row) * H_ + h) * HD_ + dt * 16 + lm] = (_Float16)o;
        }
    }
}

// ---------------------------------------------------------------------------
// LayerNorm over D per row; block = 256 threads, 4 elems/thread
// ---------------------------------------------------------------------------
__launch_bounds__(256)
__global__ void psa_ln(const float* __restrict__ y, const float* __restrict__ g,
                       const float* __restrict__ bb, float* __restrict__ out) {
    __shared__ float s1[256], s2[256];
    const int row = blockIdx.x;
    const int tid = threadIdx.x;
    const float* yr = y + (size_t)row * D_;
    float vals[4];
    float a = 0.f, sq = 0.f;
#pragma unroll
    for (int i = 0; i < 4; ++i) {
        float v = yr[tid + i * 256];
        vals[i] = v;
        a += v; sq += v * v;
    }
    s1[tid] = a; s2[tid] = sq;
    __syncthreads();
    for (int st = 128; st > 0; st >>= 1) {
        if (tid < st) { s1[tid] += s1[tid + st]; s2[tid] += s2[tid + st]; }
        __syncthreads();
    }
    const float mu  = s1[0] * (1.0f / D_);
    const float var = s2[0] * (1.0f / D_) - mu * mu;
    const float rs  = rsqrtf(var + 1e-12f);
#pragma unroll
    for (int i = 0; i < 4; ++i) {
        const int n = tid + i * 256;
        out[(size_t)row * D_ + n] = (vals[i] - mu) * rs * g[n] + bb[n];
    }
}

// ---------------------------------------------------------------------------
extern "C" void kernel_launch(void* const* d_in, const int* in_sizes, int n_in,
                              void* d_out, int out_size, void* d_ws, size_t ws_size,
                              hipStream_t stream) {
    (void)in_sizes; (void)n_in; (void)out_size; (void)ws_size;
    const float* hidden = (const float*)d_in[0];
    const float* mask   = (const float*)d_in[1];
    const float* cosp   = (const float*)d_in[2];
    const float* sinp   = (const float*)d_in[3];
    const float* mag    = (const float*)d_in[4];
    const float* Wq     = (const float*)d_in[5];
    const float* bq     = (const float*)d_in[6];
    const float* Wk     = (const float*)d_in[7];
    const float* bk     = (const float*)d_in[8];
    const float* Wv     = (const float*)d_in[9];
    const float* bv     = (const float*)d_in[10];
    const float* Wo     = (const float*)d_in[11];
    const float* bo     = (const float*)d_in[12];
    const float* Wp     = (const float*)d_in[13];
    const float* bp     = (const float*)d_in[14];
    const float* ln_g   = (const float*)d_in[15];
    const float* ln_b   = (const float*)d_in[16];
    float* out = (float*)d_out;

    char* ws = (char*)d_ws;
    size_t off = 0;
    auto nxt = [&](size_t bytes) -> void* {
        void* p = ws + off;
        off += (bytes + 255) & ~(size_t)255;
        return p;
    };
    _Float16* Xh   = (_Float16*)nxt((size_t)BL_ * D_ * 2);     // 16 MB
    _Float16* Wqt  = (_Float16*)nxt((size_t)D_ * D_ * 2);      // transposed f16 weights
    _Float16* Wkt  = (_Float16*)nxt((size_t)D_ * D_ * 2);
    _Float16* Wvt  = (_Float16*)nxt((size_t)D_ * D_ * 2);
    _Float16* Wot  = (_Float16*)nxt((size_t)D_ * D_ * 2);
    _Float16* Qh   = (_Float16*)nxt((size_t)BL_ * D_ * 2);     // [B,H,L,HD]
    _Float16* Kh   = (_Float16*)nxt((size_t)BL_ * D_ * 2);     // [B,H,L,HD]
    _Float16* Vrh  = (_Float16*)nxt((size_t)BL_ * D_ * 2);     // [B,H,HD,L]
    _Float16* Vih  = (_Float16*)nxt((size_t)BL_ * D_ * 2);     // [B,H,HD,L]
    _Float16* ctxh = (_Float16*)nxt((size_t)BL_ * D_ * 2);     // [B,L,H*HD]
    float* pcb  = (float*)nxt((size_t)BHL_ * 4);
    float* psb  = (float*)nxt((size_t)BHL_ * 4);
    float* pmc  = (float*)nxt((size_t)BHL_ * 4);
    float* pms  = (float*)nxt((size_t)BHL_ * 4);
    float* ybuf = (float*)nxt((size_t)BL_ * D_ * 4);           // 32 MB

    // 1) converts (weights transposed)
    {
        int nX = BL_ * D_;                  // 8388608
        psa_cvt_f16<<<nX / 4 / 256, 256, 0, stream>>>(hidden, Xh, nX);
        int nW4 = D_ * D_ / 4;              // 262144 threads
        psa_cvt_tr_f16<<<nW4 / 256, 256, 0, stream>>>(Wq, Wqt);
        psa_cvt_tr_f16<<<nW4 / 256, 256, 0, stream>>>(Wk, Wkt);
        psa_cvt_tr_f16<<<nW4 / 256, 256, 0, stream>>>(Wv, Wvt);
        psa_cvt_tr_f16<<<nW4 / 256, 256, 0, stream>>>(Wo, Wot);
    }
    // 2) phase projections
    psa_phase<<<BHL_ / 256, 256, 0, stream>>>(cosp, sinp, mag, Wp, bp, pcb, psb, pmc, pms);

    // 3) Q/K/V GEMMs (modes 0..2), block tile 128x128
    psa_gemm<<<dim3(D_ / 128, BL_ / 128, 3), 256, 0, stream>>>(
        Xh, Wqt, Wkt, Wvt, Wot, bq, bk, bv, bo, hidden, pmc, pms,
        Qh, Kh, Vrh, Vih, ybuf, 0);

    // 4) attention
    psa_attn<<<dim3(L_ / 128, B_ * H_), 256, 0, stream>>>(
        Qh, Kh, Vrh, Vih, mask, pcb, psb, ctxh);

    // 5) output projection + residual (mode 3)
    psa_gemm<<<dim3(D_ / 128, BL_ / 128, 1), 256, 0, stream>>>(
        ctxh, Wqt, Wkt, Wvt, Wot, bq, bk, bv, bo, hidden, pmc, pms,
        Qh, Kh, Vrh, Vih, ybuf, 3);

    // 6) layernorm
    psa_ln<<<BL_, 256, 0, stream>>>(ybuf, ln_g, ln_b, out);
}
